// RTAttention_11106785427561
// MI455X (gfx1250) — matrix operand order repaired and checked by reading, not compiled, not run on hardware
//
#include <hip/hip_runtime.h>
#include <hip/hip_bf16.h>
#include <math.h>

// ---------------------------------------------------------------------------
// RTAttention fused kernel for MI455X (gfx1250, wave32, WMMA 16x16x32 bf16)
//
// Cost model: edge projections = 2*384*384 rows x (64 -> 128) x 3 = 14.5 GFLOP
// (everything else <1%). dense_adj = 75.5 MB -> 3.2us @ 23.3 TB/s, so the
// edge GEMM is compute bound -> bf16 WMMA (16x16x32) with f32 accumulation.
// The j-loop is software-pipelined 2 deep so XDL WMMAs co-execute with the
// online-softmax VALU/exp work; f32->bf16 is pair-packed through v_perm_b32.
// ---------------------------------------------------------------------------

typedef __attribute__((ext_vector_type(16))) __bf16 v16bf;
typedef __attribute__((ext_vector_type(8)))  float  v8f;

#define B_   2
#define N_   384
#define DM   128           // NODE_DIM
#define H_   8             // HEADS
#define D_   16            // per-head dim
#define E_   64            // EDGE_DIM
#define JT   (N_ / 16)     // 24 key tiles
#define ROWE 72            // padded LDS row (64 bf16 + 8 pad) -> 144B, 16B aligned
#define SCALE_ 0.08838834764831845f                  // 1/sqrt(128)
#define C2_    (SCALE_ * 1.4426950408889634f)        // SCALE * log2(e)

// RNE-round two f32 to bf16 and pack into one dword: {hi.bf16, lo.bf16}.
// v_bfe + v_add3 per element, then one v_perm_b32 grabs both high halves.
__device__ __forceinline__ unsigned f2bf_pk(float lo, float hi) {
    unsigned u0 = __builtin_bit_cast(unsigned, lo);
    unsigned u1 = __builtin_bit_cast(unsigned, hi);
    u0 += 0x7FFFu + ((u0 >> 16) & 1u);
    u1 += 0x7FFFu + ((u1 >> 16) & 1u);
    // result bytes (MSB..LSB) = u1.b3, u1.b2, u0.b3, u0.b2
    return __builtin_amdgcn_perm(u1, u0, 0x07060302u);
}

struct U32B  { uint4 lo, hi; };
struct U8x   { unsigned u[8]; };
__device__ __forceinline__ v16bf make_frag(uint4 lo, uint4 hi) {
    U32B t{lo, hi};
    return __builtin_bit_cast(v16bf, t);   // e0..7 = lo bytes, e8..15 = hi bytes
}

// ---------------------------------------------------------------------------
// Kernel 1: node projections qn/kn/vn = x @ W.T  (exact f32; 75 MFLOP total)
// ---------------------------------------------------------------------------
__global__ void node_proj_kernel(const float* __restrict__ x,
                                 const float* __restrict__ wq,
                                 const float* __restrict__ wk,
                                 const float* __restrict__ wv,
                                 float* __restrict__ qn,
                                 float* __restrict__ kn,
                                 float* __restrict__ vn) {
    __shared__ float xs[DM];
    const int row = blockIdx.x;      // 0 .. B*N-1
    const int c   = threadIdx.x;     // output channel 0..127
    xs[c] = x[(size_t)row * DM + c];
    __syncthreads();
    const float* wqr = wq + (size_t)c * DM;
    const float* wkr = wk + (size_t)c * DM;
    const float* wvr = wv + (size_t)c * DM;
    float aq = 0.f, ak = 0.f, av = 0.f;
#pragma unroll 8
    for (int k = 0; k < DM; ++k) {
        const float xv = xs[k];
        aq = fmaf(xv, wqr[k], aq);
        ak = fmaf(xv, wkr[k], ak);
        av = fmaf(xv, wvr[k], av);
    }
    qn[(size_t)row * DM + c] = aq;
    kn[(size_t)row * DM + c] = ak;
    vn[(size_t)row * DM + c] = av;
}

// per-tile pipeline stage state: 3 WMMA accumulators + prefetched kn/vn gathers
struct TileAcc {
    v8f   cq, ck, cv;
    float kv[8], vv[8];
};

// ---------------------------------------------------------------------------
// Kernel 2: fused edge-projection + attention. One WG per (b, i) query row.
// 8 waves = 8 heads; per j-tile each wave issues 6 v_wmma_f32_16x16x32_bf16
// (qe/ke/ve, K=64 split into 2), then an online softmax over the 16 keys.
// ---------------------------------------------------------------------------
__global__ __launch_bounds__(256, 1)
void edge_attn_kernel(const float* __restrict__ adj,
                      const float* __restrict__ weq,
                      const float* __restrict__ wek,
                      const float* __restrict__ wev,
                      const float* __restrict__ qn,
                      const float* __restrict__ kn,
                      const float* __restrict__ vn,
                      float* __restrict__ out) {
    __shared__ __bf16 adj_s[N_ * ROWE];            // 54 KB: adj[b,i,:,:] as bf16

    const int blk  = blockIdx.x;                   // 0 .. B*N-1
    const int b    = blk / N_;
    const int i    = blk - b * N_;
    const int tid  = threadIdx.x;
    const int h    = tid >> 5;                     // wave id == head
    const int lane = tid & 31;
    const int n    = lane & 15;                    // WMMA column = per-head d
    const int half = lane >> 4;                    // lane half (K / M split)

    // ---- stage dense_adj[b,i,:,:] (384x64 f32) into LDS as bf16, coalesced.
    // float4 global load -> 2 packed dwords -> single 8-byte LDS store.
    const float* arow = adj + (((size_t)b * N_ + i) * N_) * E_;
    for (int idx4 = tid; idx4 < (N_ * E_) / 4; idx4 += 256) {
        const int base = idx4 * 4;
        const int j = base >> 6;                   // key row
        const int k = base & 63;                   // edge channel
        const float4 f = *reinterpret_cast<const float4*>(arow + base);
        uint2 p;
        p.x = f2bf_pk(f.x, f.y);
        p.y = f2bf_pk(f.z, f.w);
        *reinterpret_cast<uint2*>(&adj_s[j * ROWE + k]) = p;   // 8B aligned
    }

    // ---- B fragments (K x 16 slice of W.T for this head), K=64 -> two frags.
    // B layout: lane = column n, half selects K block of 16; elements = K-contig.
    auto load_bfrag = [&](const float* wrow) -> v16bf {
        U8x t;
#pragma unroll
        for (int p = 0; p < 8; ++p) t.u[p] = f2bf_pk(wrow[2 * p], wrow[2 * p + 1]);
        return __builtin_bit_cast(v16bf, t);
    };
    const float* wqrow = weq + ((size_t)(h * D_ + n)) * E_ + half * 16;
    const float* wkrow = wek + ((size_t)(h * D_ + n)) * E_ + half * 16;
    const float* wvrow = wev + ((size_t)(h * D_ + n)) * E_ + half * 16;
    const v16bf bq0 = load_bfrag(wqrow);
    const v16bf bq1 = load_bfrag(wqrow + 32);
    const v16bf bk0 = load_bfrag(wkrow);
    const v16bf bk1 = load_bfrag(wkrow + 32);
    const v16bf bv0 = load_bfrag(wvrow);
    const v16bf bv1 = load_bfrag(wvrow + 32);

    // per-lane broadcasts: qn depends only on (i, h, d)=lane; kn/vn gathered per key
    const float  qn_l = qn[(((size_t)b * N_ + i)) * DM + h * D_ + n];
    const float* knb  = kn + (size_t)b * N_ * DM + h * D_ + n;
    const float* vnb  = vn + (size_t)b * N_ * DM + h * D_ + n;

    __syncthreads();

    float m_run = -INFINITY, l_run = 0.f, acc = 0.f;   // online softmax state
    const uint4* lds4 = reinterpret_cast<const uint4*>(adj_s);   // ROWE/8 = 9 uint4/row

    // ---- pipeline stage A: A-frags from LDS, 6 WMMAs, prefetch kn/vn gathers
    auto compute_tile = [&](int jt, TileAcc& T) {
        // A frag uses contiguous k order (as B): row M = lane&15; lane<16: K in {b..b+15};
        // lane>=16: K in {b+16..b+31}. Two 16-byte runs per fragment.
        const uint4* prow = lds4 + (size_t)(jt * 16 + n) * (ROWE / 8);
        const v16bf a0 = make_frag(prow[2 * half],     prow[2 * half + 1]);   // K  0..31
        const v16bf a1 = make_frag(prow[4 + 2 * half], prow[5 + 2 * half]);   // K 32..63
        v8f cq = {}; v8f ck = {}; v8f cv = {};
        cq = __builtin_amdgcn_wmma_f32_16x16x32_bf16(false, a0, false, bq0, (short)0, cq, false, false);
        cq = __builtin_amdgcn_wmma_f32_16x16x32_bf16(false, a1, false, bq1, (short)0, cq, false, false);
        ck = __builtin_amdgcn_wmma_f32_16x16x32_bf16(false, a0, false, bk0, (short)0, ck, false, false);
        ck = __builtin_amdgcn_wmma_f32_16x16x32_bf16(false, a1, false, bk1, (short)0, ck, false, false);
        cv = __builtin_amdgcn_wmma_f32_16x16x32_bf16(false, a0, false, bv0, (short)0, cv, false, false);
        cv = __builtin_amdgcn_wmma_f32_16x16x32_bf16(false, a1, false, bv1, (short)0, cv, false, false);
        T.cq = cq; T.ck = ck; T.cv = cv;
#pragma unroll
        for (int r = 0; r < 8; ++r) {               // row m = r + 8*half
            const int j = jt * 16 + r + half * 8;
            T.kv[r] = knb[(size_t)j * DM];
            T.vv[r] = vnb[(size_t)j * DM];
        }
    };

    // ---- pipeline stage B: scores, online softmax, PV accumulate
    auto process_tile = [&](TileAcc& T) {
        float s[8];
#pragma unroll
        for (int r = 0; r < 8; ++r) {
            float t = (T.cq[r] + qn_l) * (T.ck[r] + T.kv[r]);
            t += __shfl_xor(t, 1, 32);
            t += __shfl_xor(t, 2, 32);
            t += __shfl_xor(t, 4, 32);
            t += __shfl_xor(t, 8, 32);              // all 16 lanes of half hold sum
            s[r] = t;                               // raw dot; scale folded into exp2
        }
        float tm = s[0];
#pragma unroll
        for (int r = 1; r < 8; ++r) tm = fmaxf(tm, s[r]);
        tm = fmaxf(tm, __shfl_xor(tm, 16, 32));     // max over all 16 keys in tile
        const float m_new = fmaxf(m_run, tm);
        const float alpha = __builtin_amdgcn_exp2f(C2_ * (m_run - m_new)); // 0 on first tile
        l_run *= alpha;
        acc   *= alpha;
#pragma unroll
        for (int r = 0; r < 8; ++r) {
            const float p = __builtin_amdgcn_exp2f(C2_ * (s[r] - m_new));
            l_run += p;                              // identical across lanes of half
            acc = fmaf(p, T.cv[r] + T.vv[r], acc);   // per-lane: d = lane&15
        }
        m_run = m_new;
    };

    // ---- 2-deep software pipeline over the 24 key tiles (JT is even)
    TileAcc T0, T1;
    compute_tile(0, T0);
    for (int jt = 0; jt < JT; jt += 2) {
        compute_tile(jt + 1, T1);      // WMMAs for jt+1 in flight ...
        process_tile(T0);              // ... while softmax of jt runs on VALU
        if (jt + 2 < JT) compute_tile(jt + 2, T0);
        process_tile(T1);
    }

    // combine the two half-wave partial sums (rows 0..7 / 8..15 interleave)
    acc   += __shfl_xor(acc, 16, 32);
    l_run += __shfl_xor(l_run, 16, 32);
    if (half == 0) {
        // reference output transpose: (B,N,H,d) -> (B,N,d,H): channel = d*H + h
        out[(((size_t)b * N_ + i)) * DM + n * H_ + h] = acc / l_run;
    }
}

// ---------------------------------------------------------------------------
extern "C" void kernel_launch(void* const* d_in, const int* in_sizes, int n_in,
                              void* d_out, int out_size, void* d_ws, size_t ws_size,
                              hipStream_t stream) {
    const float* x   = (const float*)d_in[0];
    const float* adj = (const float*)d_in[1];
    const float* wnq = (const float*)d_in[2];
    const float* wnk = (const float*)d_in[3];
    const float* wnv = (const float*)d_in[4];
    const float* weq = (const float*)d_in[5];
    const float* wek = (const float*)d_in[6];
    const float* wev = (const float*)d_in[7];
    float* out = (float*)d_out;

    float* qn = (float*)d_ws;                 // 3 x 768 x 128 f32 = 1.18 MB scratch
    float* kn = qn + (size_t)B_ * N_ * DM;
    float* vn = kn + (size_t)B_ * N_ * DM;

    node_proj_kernel<<<B_ * N_, DM, 0, stream>>>(x, wnq, wnk, wnv, qn, kn, vn);
    edge_attn_kernel<<<B_ * N_, 256, 0, stream>>>(adj, weq, wek, wev, qn, kn, vn, out);
}
